// MHSelfAttention_51488067945212
// MI455X (gfx1250) — matrix operand order repaired and checked by reading, not compiled, not run on hardware
//
#include <hip/hip_runtime.h>

typedef __attribute__((ext_vector_type(16))) _Float16 v16h;
typedef __attribute__((ext_vector_type(8)))  _Float16 v8h;
typedef __attribute__((ext_vector_type(8)))  float    v8f;

#define B_  4
#define L_  2048
#define D_  1024
#define H_  16
#define HD_ 64
#define BL_ (B_*L_)

union AFrag { v16h v; v8h h[2]; };

__device__ __forceinline__ v8f wmma_f16(v16h a, v16h b, v8f c) {
  // D = A(16x32 f16) * B(32x16 f16) + C(16x16 f32)
  return __builtin_amdgcn_wmma_f32_16x16x32_f16(false, a, false, b, (short)0, c,
                                                false, false);
}

// ---------------------------------------------------------------- convert
__global__ void cvt_f32_f16(const float* __restrict__ s,
                            _Float16* __restrict__ d, int n) {
  int i = blockIdx.x * 256 + threadIdx.x;
  if (i < n) d[i] = (_Float16)s[i];
}

// ------------------------------------------------- fused QKV + bias + RoPE
// grid: (BL/32, 12), block 128 (4 waves). Each wave: 32 rows x 64 features
// (one head of q, k or v). wave column wc in [0,48): seg = wc>>4, h = wc&15.
__global__ void __launch_bounds__(128)
qkv_rope(const _Float16* __restrict__ xh,
         const _Float16* __restrict__ wqh, const _Float16* __restrict__ wkh,
         const _Float16* __restrict__ wvh,
         const float* __restrict__ bq, const float* __restrict__ bk,
         const float* __restrict__ bv,
         _Float16* __restrict__ qws, _Float16* __restrict__ kws,
         _Float16* __restrict__ vtws)
{
  const int lane = threadIdx.x & 31, wid = threadIdx.x >> 5;
  const int lm = lane & 15, hf = lane >> 4;
  const int row0 = blockIdx.x * 32;
  const int wc = blockIdx.y * 4 + wid;       // 0..47
  const int seg = wc >> 4;                   // 0=q 1=k 2=v
  const int h = wc & 15;
  const _Float16* W   = (seg == 0) ? wqh : (seg == 1) ? wkh : wvh;
  const float*    bia = (seg == 0) ? bq  : (seg == 1) ? bk  : bv;

  v8f acc[2][4] = {};
#pragma unroll 1                             // keep rolled: avoid VGPR spills
  for (int kc = 0; kc < 32; ++kc) {          // K = 1024 in chunks of 32
    AFrag a[2];
#pragma unroll
    for (int mt = 0; mt < 2; ++mt) {
      const _Float16* xr =
          xh + ((size_t)(row0 + mt * 16 + lm)) * D_ + kc * 32 + hf * 16;
      a[mt].h[0] = *(const v8h*)xr;
      a[mt].h[1] = *(const v8h*)(xr + 8);
      __builtin_prefetch(xr + 32, 0, 3);
    }
#pragma unroll
    for (int t = 0; t < 4; ++t) {
      const _Float16* wr = W + ((size_t)(h * HD_ + t * 16 + lm)) * D_ +
                           kc * 32 + hf * 16;
      v16h bf = *(const v16h*)wr;
      __builtin_prefetch(wr + 32, 0, 3);
#pragma unroll
      for (int mt = 0; mt < 2; ++mt)
        acc[mt][t] = wmma_f16(a[mt].v, bf, acc[mt][t]);
    }
  }
#pragma unroll
  for (int t = 0; t < 4; ++t) {
    float bb = bia[h * HD_ + t * 16 + lm];
#pragma unroll
    for (int mt = 0; mt < 2; ++mt)
#pragma unroll
      for (int r = 0; r < 8; ++r) acc[mt][t][r] += bb;
  }

  const int b = row0 >> 11;
  const int pos0 = row0 & (L_ - 1);

  if (seg < 2) {
    // RoPE: d = t*16+lm; pairs (d, d+-32) are acc[t] <-> acc[t^2], same lane.
    const float c0f = -9.210340371976184f / 32.0f;   // -ln(10000)/32
    float if0 = __expf((float)lm * c0f);
    float if1 = __expf((float)(lm + 16) * c0f);
#pragma unroll
    for (int mt = 0; mt < 2; ++mt)
#pragma unroll
      for (int r = 0; r < 8; ++r) {
        float pf = (float)(pos0 + mt * 16 + r + 8 * hf);
        float a0 = pf * if0, a1 = pf * if1;
        float c0 = cosf(a0), s0 = sinf(a0);
        float c1 = cosf(a1), s1 = sinf(a1);
        float v0 = acc[mt][0][r], v1 = acc[mt][1][r];
        float v2 = acc[mt][2][r], v3 = acc[mt][3][r];
        acc[mt][0][r] = v0 * c0 - v2 * s0;
        acc[mt][1][r] = v1 * c1 - v3 * s1;
        acc[mt][2][r] = v2 * c0 + v0 * s0;
        acc[mt][3][r] = v3 * c1 + v1 * s1;
      }
    _Float16* dst = (seg == 0) ? qws : kws;   // [B,H,L,Hd]
#pragma unroll
    for (int mt = 0; mt < 2; ++mt)
#pragma unroll
      for (int t = 0; t < 4; ++t)
#pragma unroll
        for (int r = 0; r < 8; ++r)
          dst[((size_t)((b * H_ + h) * L_ + pos0 + mt * 16 + r + 8 * hf)) *
                  HD_ + t * 16 + lm] = (_Float16)acc[mt][t][r];
  } else {
    // V stored transposed [B,H,Hd,L]; per (lane,mt,t): 8 consecutive pos.
#pragma unroll
    for (int mt = 0; mt < 2; ++mt)
#pragma unroll
      for (int t = 0; t < 4; ++t) {
        v8h pk;
#pragma unroll
        for (int r = 0; r < 8; ++r) pk[r] = (_Float16)acc[mt][t][r];
        *(v8h*)(vtws + ((size_t)((b * H_ + h) * HD_ + t * 16 + lm)) * L_ +
                pos0 + mt * 16 + 8 * hf) = pk;
      }
  }
}

// ------------------------------------------------------- flash attention
// grid: 1024 blocks x 128 threads; each wave: one (b,h), 2 query tiles.
__global__ void __launch_bounds__(128)
flash_attn(const _Float16* __restrict__ qws, const _Float16* __restrict__ kws,
           const _Float16* __restrict__ vtws,
           const unsigned char* __restrict__ mask,
           _Float16* __restrict__ aws)
{
  __shared__ __align__(16) _Float16 lp[4][16 * 40];  // per-wave P tile, pitch 40
  const int lane = threadIdx.x & 31, wid = threadIdx.x >> 5;
  const int lm = lane & 15, hf = lane >> 4;
  const int gw = blockIdx.x * 4 + wid;       // 0..4095
  const int bh = gw >> 6;                    // 64 wave-jobs per (b,h)
  const int b  = bh >> 4;
  const int h  = bh & 15;
  const int q0 = (gw & 63) * 32;

  AFrag aq[2][2];
#pragma unroll
  for (int qt = 0; qt < 2; ++qt)
#pragma unroll
    for (int kc = 0; kc < 2; ++kc) {
      const _Float16* p = qws + ((size_t)(bh * L_ + q0 + qt * 16 + lm)) * HD_ +
                          kc * 32 + hf * 16;
      aq[qt][kc].h[0] = *(const v8h*)p;
      aq[qt][kc].h[1] = *(const v8h*)(p + 8);
    }

  v8f acc[2][4] = {};
  float rm[2][8], rl[2][8];
#pragma unroll
  for (int qt = 0; qt < 2; ++qt)
#pragma unroll
    for (int r = 0; r < 8; ++r) { rm[qt][r] = -1e30f; rl[qt][r] = 0.0f; }

  const unsigned char* mrow = mask + b * L_;

#pragma unroll 1                             // keep rolled: avoid VGPR spills
  for (int n0 = 0; n0 < L_; n0 += 32) {
    v16h bk[2][2];
#pragma unroll
    for (int kt = 0; kt < 2; ++kt)
#pragma unroll
      for (int kc = 0; kc < 2; ++kc) {
        const _Float16* kr = kws +
            ((size_t)(bh * L_ + n0 + kt * 16 + lm)) * HD_ + kc * 32 + hf * 16;
        bk[kt][kc] = *(const v16h*)kr;
        __builtin_prefetch(kr + 32 * HD_, 0, 3);       // next key chunk
      }

    const bool mk0 = mrow[n0 + lm] != 0;        // lane's key column, tile 0
    const bool mk1 = mrow[n0 + 16 + lm] != 0;   // tile 1

#pragma unroll
    for (int qt = 0; qt < 2; ++qt) {
      v8f s0 = {}, s1 = {};
      s0 = wmma_f16(aq[qt][0].v, bk[0][0], s0);
      s0 = wmma_f16(aq[qt][1].v, bk[0][1], s0);
      s1 = wmma_f16(aq[qt][0].v, bk[1][0], s1);
      s1 = wmma_f16(aq[qt][1].v, bk[1][1], s1);

      float p0[8], p1[8];
#pragma unroll
      for (int r = 0; r < 8; ++r) {
        float a0 = mk0 ? -1e30f : s0[r] * 0.125f;
        float a1 = mk1 ? -1e30f : s1[r] * 0.125f;
        float mx = fmaxf(a0, a1);
        mx = fmaxf(mx, __shfl_xor(mx, 1));
        mx = fmaxf(mx, __shfl_xor(mx, 2));
        mx = fmaxf(mx, __shfl_xor(mx, 4));
        mx = fmaxf(mx, __shfl_xor(mx, 8));
        float mnew = fmaxf(rm[qt][r], mx);
        float e0 = __expf(a0 - mnew);
        float e1 = __expf(a1 - mnew);
        float sum = e0 + e1;
        sum += __shfl_xor(sum, 1);
        sum += __shfl_xor(sum, 2);
        sum += __shfl_xor(sum, 4);
        sum += __shfl_xor(sum, 8);
        float corr = __expf(rm[qt][r] - mnew);
        rl[qt][r] = rl[qt][r] * corr + sum;
        rm[qt][r] = mnew;
#pragma unroll
        for (int dt = 0; dt < 4; ++dt) acc[qt][dt][r] *= corr;
        p0[r] = e0; p1[r] = e1;
      }

      // transpose P (C-layout) -> A-layout through private LDS region
#pragma unroll
      for (int r = 0; r < 8; ++r) {
        lp[wid][(r + 8 * hf) * 40 + lm]      = (_Float16)p0[r];
        lp[wid][(r + 8 * hf) * 40 + 16 + lm] = (_Float16)p1[r];
      }
      asm volatile("s_wait_dscnt 0" ::: "memory");
      AFrag ap;
      ap.h[0] = *(const v8h*)&lp[wid][lm * 40 + hf * 16];
      ap.h[1] = *(const v8h*)&lp[wid][lm * 40 + hf * 16 + 8];

#pragma unroll
      for (int dt = 0; dt < 4; ++dt) {
        const _Float16* vr = vtws +
            ((size_t)(bh * HD_ + dt * 16 + lm)) * L_ + n0 + hf * 16;
        v16h bv = *(const v16h*)vr;
        if (qt == 0) __builtin_prefetch(vr + 32, 0, 3);
        acc[qt][dt] = wmma_f16(ap.v, bv, acc[qt][dt]);
      }
    }
  }

#pragma unroll
  for (int qt = 0; qt < 2; ++qt)
#pragma unroll
    for (int dt = 0; dt < 4; ++dt)
#pragma unroll
      for (int r = 0; r < 8; ++r) {
        float o = acc[qt][dt][r] / rl[qt][r];
        aws[((size_t)(b * L_ + q0 + qt * 16 + r + 8 * hf)) * D_ +
            h * HD_ + dt * 16 + lm] = (_Float16)o;
      }
}

// ------------------------------------------------------- output projection
// grid: (BL/32, 4), block 128. Each wave: 32 rows x 64 features.
__global__ void __launch_bounds__(128)
oproj(const _Float16* __restrict__ ah, const _Float16* __restrict__ woh,
      const float* __restrict__ bo, float* __restrict__ out)
{
  const int lane = threadIdx.x & 31, wid = threadIdx.x >> 5;
  const int lm = lane & 15, hf = lane >> 4;
  const int row0 = blockIdx.x * 32;
  const int wc = blockIdx.y * 4 + wid;     // 0..15 -> 64 features each
  v8f acc[2][4] = {};
#pragma unroll 1                           // keep rolled: avoid VGPR spills
  for (int kc = 0; kc < 32; ++kc) {
    AFrag a[2];
#pragma unroll
    for (int mt = 0; mt < 2; ++mt) {
      const _Float16* xr =
          ah + ((size_t)(row0 + mt * 16 + lm)) * D_ + kc * 32 + hf * 16;
      a[mt].h[0] = *(const v8h*)xr;
      a[mt].h[1] = *(const v8h*)(xr + 8);
      __builtin_prefetch(xr + 32, 0, 3);
    }
#pragma unroll
    for (int t = 0; t < 4; ++t) {
      const _Float16* wr = woh + ((size_t)(wc * HD_ + t * 16 + lm)) * D_ +
                           kc * 32 + hf * 16;
      v16h bf = *(const v16h*)wr;
      __builtin_prefetch(wr + 32, 0, 3);
#pragma unroll
      for (int mt = 0; mt < 2; ++mt)
        acc[mt][t] = wmma_f16(a[mt].v, bf, acc[mt][t]);
    }
  }
#pragma unroll
  for (int t = 0; t < 4; ++t) {
    float bb = bo[wc * HD_ + t * 16 + lm];
#pragma unroll
    for (int mt = 0; mt < 2; ++mt)
#pragma unroll
      for (int r = 0; r < 8; ++r)
        out[((size_t)(row0 + mt * 16 + r + 8 * hf)) * D_ + wc * HD_ +
            t * 16 + lm] = acc[mt][t][r] + bb;
  }
}

// ------------------------------------------------------------------ launch
extern "C" void kernel_launch(void* const* d_in, const int* in_sizes, int n_in,
                              void* d_out, int out_size, void* d_ws,
                              size_t ws_size, hipStream_t stream)
{
  (void)in_sizes; (void)n_in; (void)out_size; (void)ws_size;
  const float* x  = (const float*)d_in[0];
  const float* wq = (const float*)d_in[1];
  const float* bq = (const float*)d_in[2];
  const float* wk = (const float*)d_in[3];
  const float* bk = (const float*)d_in[4];
  const float* wv = (const float*)d_in[5];
  const float* bv = (const float*)d_in[6];
  const float* wo = (const float*)d_in[7];
  const float* bo = (const float*)d_in[8];
  const unsigned char* mask = (const unsigned char*)d_in[9];

  const size_t XN = (size_t)BL_ * D_;   // 8M elems
  const size_t WN = (size_t)D_ * D_;    // 1M elems
  char* ws = (char*)d_ws;               // 88 MB total (f16)
  _Float16* xh   = (_Float16*)ws;
  _Float16* wqh  = (_Float16*)(ws + 2 * XN);
  _Float16* wkh  = wqh + WN;
  _Float16* wvh  = wkh + WN;
  _Float16* woh  = wvh + WN;
  _Float16* qws  = woh + WN;
  _Float16* kws  = qws + XN;
  _Float16* vtws = kws + XN;
  _Float16* ah   = vtws + XN;

  cvt_f32_f16<<<(int)((XN + 255) / 256), 256, 0, stream>>>(x, xh, (int)XN);
  cvt_f32_f16<<<(int)((WN + 255) / 256), 256, 0, stream>>>(wq, wqh, (int)WN);
  cvt_f32_f16<<<(int)((WN + 255) / 256), 256, 0, stream>>>(wk, wkh, (int)WN);
  cvt_f32_f16<<<(int)((WN + 255) / 256), 256, 0, stream>>>(wv, wvh, (int)WN);
  cvt_f32_f16<<<(int)((WN + 255) / 256), 256, 0, stream>>>(wo, woh, (int)WN);

  qkv_rope<<<dim3(BL_ / 32, 12), 128, 0, stream>>>(xh, wqh, wkh, wvh,
                                                   bq, bk, bv, qws, kws, vtws);
  flash_attn<<<dim3((B_ * H_ * (L_ / 32)) / 4), 128, 0, stream>>>(
      qws, kws, vtws, mask, ah);
  oproj<<<dim3(BL_ / 32, 4), 128, 0, stream>>>(ah, woh, bo, (float*)d_out);
}
